// DiscoveryNet_21629455303242
// MI455X (gfx1250) — matrix operand order repaired and checked by reading, not compiled, not run on hardware
//
#include <hip/hip_runtime.h>
#include <hip/hip_bf16.h>
#include <math.h>

typedef __attribute__((ext_vector_type(16))) _Float16 v16h;
typedef __attribute__((ext_vector_type(8)))  float    v8f;

#define HIDDEN 128
#define NPTS   1024
#define NBATCH 4
#define MIN_DIST 0.01f
#define TWO_LOG2E 2.8853900817779268f   // 2*log2(e)

#if __has_builtin(__builtin_amdgcn_sched_barrier)
#define SCHED_FENCE() __builtin_amdgcn_sched_barrier(0)
#else
#define SCHED_FENCE() asm volatile("" ::: "memory")
#endif

// tanh from a PRE-SCALED argument s = 2*log2(e)*x:
// tanh(x) = 1 - 2*rcp(exp2(s) + 1). Branchless; exact +/-1 saturation.
__device__ __forceinline__ float tanh_exparg(float s) {
    float e = __builtin_amdgcn_exp2f(s);
    return __builtin_fmaf(-2.0f, __builtin_amdgcn_rcpf(e + 1.0f), 1.0f);
}

// Each block computes out[b][i][0..2] = sum_j mag(i,j) * (pos_i - pos_j)/dist_clip
// 256 threads = 8 waves; wave w handles j in [w*128, w*128+128) as 8 WMMA M-tiles.
__global__ __launch_bounds__(256) void
discovery_force_kernel(const float* __restrict__ pos,   // (B,N,3)
                       const float* __restrict__ W1,    // (3,128)
                       const float* __restrict__ b1,    // (128)
                       const float* __restrict__ W2,    // (128,128)
                       const float* __restrict__ b2,    // (128)
                       const float* __restrict__ W3,    // (128,1)
                       const float* __restrict__ b3,    // (1)
                       float* __restrict__ out)         // (B,N,3)
{
    __shared__ __align__(32) _Float16 sW2T[HIDDEN * HIDDEN]; // [col][k], 32KB
    __shared__ float sPos[NPTS * 3];                         // 12KB
    __shared__ __align__(16) float sL1[HIDDEN * 4];  // 2log2e*{W1r0,W1r1,W1r2,b1}[k]
    __shared__ __align__(8)  float sL23[HIDDEN * 2]; // {2log2e*b2, W3}[col]
    __shared__ float sWsum[8][3];

    const int tid  = threadIdx.x;
    const int bi   = blockIdx.x;          // b*N + i
    const int b    = bi >> 10;
    const int i    = bi & (NPTS - 1);
    const int wave = tid >> 5;
    const int lane = tid & 31;
    const int laneM = lane & 15;          // row (A) or column (B/C) within tile
    const int half  = lane >> 4;

    // ---- cooperative staging into LDS ----
    const float* posB = pos + (size_t)b * NPTS * 3;
    for (int idx = tid; idx < NPTS * 3; idx += 256) sPos[idx] = posB[idx];
    // W2 (k,col) row-major -> f16 transposed [col][k]
    for (int idx = tid; idx < HIDDEN * HIDDEN; idx += 256) {
        int k = idx >> 7, c = idx & 127;
        sW2T[c * HIDDEN + k] = (_Float16)W2[idx];
    }
    if (tid < HIDDEN) {
        // Prescale layer-1 weights/bias and layer-2 bias by 2*log2(e) so the
        // tanh exp2 argument needs no extra multiply.
        sL1[tid * 4 + 0] = W1[tid]              * TWO_LOG2E;
        sL1[tid * 4 + 1] = W1[HIDDEN + tid]     * TWO_LOG2E;
        sL1[tid * 4 + 2] = W1[2 * HIDDEN + tid] * TWO_LOG2E;
        sL1[tid * 4 + 3] = b1[tid]              * TWO_LOG2E;
        sL23[tid * 2 + 0] = b2[tid] * TWO_LOG2E;
        sL23[tid * 2 + 1] = W3[tid];
    }
    __syncthreads();

    const float pix = sPos[i * 3 + 0];
    const float piy = sPos[i * 3 + 1];
    const float piz = sPos[i * 3 + 2];
    const float b3v = b3[0];

    float fx = 0.0f, fy = 0.0f, fz = 0.0f;

    #pragma unroll 1
    for (int m = 0; m < 8; ++m) {
        const int rowbase = wave * 128 + m * 16;

        // ---- layer 1: feat -> tanh -> A fragments (16x32 f16 per K-block) ----
        // 16-bit A layout: lanes 0-15 row=laneM, VGPR v: K={2v,2v+1} (v<4) or
        // {16+2(v-4),..} (v>=4); lanes 16-31 same rows, K offset +8.
        const int jr = rowbase + laneM;
        float dx = pix - sPos[jr * 3 + 0];
        float dy = piy - sPos[jr * 3 + 1];
        float dz = piz - sPos[jr * 3 + 2];
        float d2 = dx * dx + dy * dy + dz * dz;
        float dist = sqrtf(d2);
        float dc = fmaxf(dist, MIN_DIST);
        float ir = __builtin_amdgcn_rcpf(dc);   // feeds f16 features only
        float f0 = dist, f1 = ir, f2 = ir * ir;

        v16h A[4];
        #pragma unroll
        for (int kb = 0; kb < 4; ++kb) {
            #pragma unroll
            for (int v = 0; v < 8; ++v) {
                int base = (v < 4) ? (half * 8 + 2 * v)
                                   : (16 + half * 8 + 2 * (v - 4));
                int k0 = kb * 32 + base;
                float4 wa = *(const float4*)&sL1[k0 * 4];
                float4 wb = *(const float4*)&sL1[(k0 + 1) * 4];
                float h0 = tanh_exparg(f0 * wa.x + f1 * wa.y + f2 * wa.z + wa.w);
                float h1 = tanh_exparg(f0 * wb.x + f1 * wb.y + f2 * wb.z + wb.w);
                A[kb][2 * v]     = (_Float16)h0;
                A[kb][2 * v + 1] = (_Float16)h1;
            }
            SCHED_FENCE();   // cap live tanh chains to one kb-group (16)
        }

        float pm[8];
        #pragma unroll
        for (int v = 0; v < 8; ++v) pm[v] = 0.0f;

        // ---- layer 2 (WMMA over K=128) + layer 3 partial dot ----
        // Rolled: only one set of B fragments live -> no spills.
        #pragma unroll 1
        for (int nb = 0; nb < 8; ++nb) {
            const int col = nb * 16 + laneM;
            // B fragment: fixed column, K = 8*half+s (s<8), 16+8*half+(s-8) (s>=8)
            v16h Bf[4];
            #pragma unroll
            for (int kb = 0; kb < 4; ++kb) {
                #pragma unroll
                for (int s = 0; s < 16; ++s)
                    Bf[kb][s] = sW2T[col * HIDDEN + kb * 32 + ((s < 8) ? (half * 8 + s) : (16 + half * 8 + (s - 8)))];
            }
            float2 bw = *(const float2*)&sL23[col * 2]; // {2log2e*b2[col], W3[col]}

            v8f acc = {};   // C = inline 0; bias folded into exp2 argument below
            acc = __builtin_amdgcn_wmma_f32_16x16x32_f16(false, A[0], false, Bf[0],
                                                         (short)0, acc, false, false);
            acc = __builtin_amdgcn_wmma_f32_16x16x32_f16(false, A[1], false, Bf[1],
                                                         (short)0, acc, false, false);
            acc = __builtin_amdgcn_wmma_f32_16x16x32_f16(false, A[2], false, Bf[2],
                                                         (short)0, acc, false, false);
            acc = __builtin_amdgcn_wmma_f32_16x16x32_f16(false, A[3], false, Bf[3],
                                                         (short)0, acc, false, false);

            #pragma unroll
            for (int v = 0; v < 8; ++v) {
                // s = 2log2e*(h2 + b2[col]) via single fma against prescaled bias
                float s = __builtin_fmaf(acc[v], TWO_LOG2E, bw.x);
                pm[v] += tanh_exparg(s) * bw.y; // row v+8*half, this lane's col
            }
        }

        // ---- finalize: reduce over 16 columns per row, accumulate force ----
        #pragma unroll
        for (int v = 0; v < 8; ++v) {
            float r = pm[v];
            r += __shfl_xor(r, 1, 32);
            r += __shfl_xor(r, 2, 32);
            r += __shfl_xor(r, 4, 32);
            r += __shfl_xor(r, 8, 32);
            if (laneM == v) {
                int j = rowbase + v + 8 * half;
                if (j != i) {
                    float ddx = pix - sPos[j * 3 + 0];
                    float ddy = piy - sPos[j * 3 + 1];
                    float ddz = piz - sPos[j * 3 + 2];
                    float dd2 = ddx * ddx + ddy * ddy + ddz * ddz;
                    float dsj = sqrtf(dd2);
                    float dcj = fmaxf(dsj, MIN_DIST);
                    float iij = 1.0f / dcj;      // exact path for final force
                    float mag = r + b3v;
                    fx += mag * ddx * iij;
                    fy += mag * ddy * iij;
                    fz += mag * ddz * iij;
                }
            }
        }
    }

    // ---- wave reduction then block reduction ----
    #pragma unroll
    for (int s = 1; s < 32; s <<= 1) {
        fx += __shfl_xor(fx, s, 32);
        fy += __shfl_xor(fy, s, 32);
        fz += __shfl_xor(fz, s, 32);
    }
    if (lane == 0) {
        sWsum[wave][0] = fx;
        sWsum[wave][1] = fy;
        sWsum[wave][2] = fz;
    }
    __syncthreads();
    if (tid < 3) {
        float s = 0.0f;
        #pragma unroll
        for (int w = 0; w < 8; ++w) s += sWsum[w][tid];
        out[(size_t)bi * 3 + tid] = s;
    }
}

extern "C" void kernel_launch(void* const* d_in, const int* in_sizes, int n_in,
                              void* d_out, int out_size, void* d_ws, size_t ws_size,
                              hipStream_t stream) {
    (void)in_sizes; (void)n_in; (void)d_ws; (void)ws_size; (void)out_size;
    const float* pos = (const float*)d_in[0];
    const float* W1  = (const float*)d_in[1];
    const float* b1  = (const float*)d_in[2];
    const float* W2  = (const float*)d_in[3];
    const float* b2  = (const float*)d_in[4];
    const float* W3  = (const float*)d_in[5];
    const float* b3  = (const float*)d_in[6];
    float* out = (float*)d_out;

    dim3 grid(NBATCH * NPTS);   // 4096 blocks, one per (b, i)
    dim3 block(256);            // 8 wave32
    discovery_force_kernel<<<grid, block, 0, stream>>>(pos, W1, b1, W2, b2, W3, b3, out);
}
